// SimpleConditionalNCA_70712341561437
// MI455X (gfx1250) — matrix operand order repaired and checked by reading, not compiled, not run on hardware
//
#include <hip/hip_runtime.h>
#include <stdint.h>

typedef __attribute__((ext_vector_type(16))) _Float16 v16h;
typedef __attribute__((ext_vector_type(8)))  float    v8f;

#define GC    12
#define PCH   36
#define HID   64
#define CDIM  32
#define BATCH 64
#define HH    128
#define WW    128
#define NSTEPS 32

#define TW 32
#define TH 8
#define TPX (TW*TH)          // 256 pixels per block
#define KP 64                // padded K for GEMM1 (36 -> 64)
#define PSTRIDE 66           // f16 row stride (odd dword count -> conflict-free)
#define WSTRIDE 66
#define USTRIDE 17           // f32 row stride for upd

// ---- gfx1250 async global->LDS path (guarded; falls back to plain loads) ----
#if defined(__HIP_DEVICE_COMPILE__) && __has_builtin(__builtin_amdgcn_global_load_async_to_lds_b32)
#define NCA_ASYNC_LDS 1
typedef __attribute__((address_space(1))) int gint_t;   // global int
typedef __attribute__((address_space(3))) int lint_t;   // LDS int
__device__ __forceinline__ void async_g2l_b32(const void* g, void* l) {
    __builtin_amdgcn_global_load_async_to_lds_b32(
        (gint_t*)(unsigned long long)(uintptr_t)g,
        (lint_t*)(unsigned)(uintptr_t)l, 0, 0);
}
__device__ __forceinline__ void async_wait0() {
#if __has_builtin(__builtin_amdgcn_s_wait_asynccnt)
    __builtin_amdgcn_s_wait_asynccnt(0);
#else
    asm volatile("s_wait_asynccnt 0" ::: "memory");
#endif
}
#endif

// ---------------------------------------------------------------- seed init
__global__ void nca_init_kernel(float* __restrict__ grid) {
    const long long n = (long long)BATCH * GC * HH * WW;
    for (long long i = (long long)blockIdx.x * blockDim.x + threadIdx.x; i < n;
         i += (long long)gridDim.x * blockDim.x) {
        int x = (int)(i % WW);
        int y = (int)((i / WW) % HH);
        int c = (int)((i / ((long long)WW * HH)) % GC);
        float v = (c >= 3 && c < 6 &&
                   y >= HH/2 - 1 && y <= HH/2 &&
                   x >= WW/2 - 1 && x <= WW/2) ? 1.0f : 0.0f;
        grid[i] = v;
    }
}

// ------------------------------------------------------------- FiLM generator
__global__ void nca_film_kernel(const int* __restrict__ labels,
                                const float* __restrict__ emb,
                                const float* __restrict__ f1w, const float* __restrict__ f1b,
                                const float* __restrict__ f2w, const float* __restrict__ f2b,
                                float* __restrict__ gb) {
    int b = blockIdx.x * blockDim.x + threadIdx.x;
    if (b >= BATCH) return;
    const float* cond = emb + labels[b] * CDIM;
    float h[CDIM];
    for (int j = 0; j < CDIM; ++j) {
        float a = f1b[j];
        for (int k = 0; k < CDIM; ++k) a += cond[k] * f1w[j * CDIM + k];
        h[j] = fmaxf(a, 0.0f);
    }
    for (int o = 0; o < 2 * GC; ++o) {
        float a = f2b[o];
        for (int j = 0; j < CDIM; ++j) a += h[j] * f2w[o * CDIM + j];
        gb[b * (2 * GC) + o] = a;
    }
}

// ---------------------------------------------------------------- NCA step
__global__ __launch_bounds__(256) void nca_step_kernel(
    const float* __restrict__ gin, float* __restrict__ gout,
    const float* __restrict__ pw, const float* __restrict__ w1,
    const float* __restrict__ b1, const float* __restrict__ w2,
    const float* __restrict__ gb,
    float* __restrict__ traj, float* __restrict__ finalrgb) {

    __shared__ float    sGrid[GC][TH + 2][TW + 2];     // halo tile (f32)
    __shared__ _Float16 sPerc[TPX][PSTRIDE];           // perceived, K padded to 64
    __shared__ _Float16 sHid [TPX][PSTRIDE];           // hidden (post ReLU), K=64
    __shared__ float    sUpd [TPX][USTRIDE];           // update, 12 valid channels
    __shared__ float    sPw[PCH * 9];
    __shared__ float    sB1[HID];
    __shared__ _Float16 sW1[HID][WSTRIDE];             // w1 as B-matrix [n][k], K padded
    __shared__ _Float16 sW2[16][WSTRIDE];              // w2 as B-matrix [n][k], N padded
    __shared__ float    sGB[2 * GC];

    const int tid  = threadIdx.x;
    const int b    = blockIdx.z;
    const int tx0  = blockIdx.x * TW;
    const int ty0  = blockIdx.y * TH;

    // ---- Phase A: grid tile + circular halo (async direct-to-LDS if available) ----
    const float* gbin = gin + (size_t)b * GC * HH * WW;
    for (int i = tid; i < GC * (TH + 2) * (TW + 2); i += 256) {
        int x = i % (TW + 2);
        int r = i / (TW + 2);
        int y = r % (TH + 2);
        int c = r / (TH + 2);
        int gx = (tx0 - 1 + x) & (WW - 1);
        int gy = (ty0 - 1 + y) & (HH - 1);
        const float* src = &gbin[(size_t)c * HH * WW + (size_t)gy * WW + gx];
#ifdef NCA_ASYNC_LDS
        async_g2l_b32(src, &sGrid[c][y][x]);
#else
        sGrid[c][y][x] = *src;
#endif
    }

    // ---- stage weights into LDS (overlaps with async tile DMA) ----
    for (int i = tid; i < PCH * 9; i += 256) sPw[i] = pw[i];
    for (int i = tid; i < HID; i += 256)     sB1[i] = b1[i];
    for (int i = tid; i < HID * KP; i += 256) {
        int n = i >> 6, k = i & 63;
        sW1[n][k] = (_Float16)((k < PCH) ? w1[n * PCH + k] : 0.0f);
    }
    for (int i = tid; i < 16 * KP; i += 256) {
        int n = i >> 6, k = i & 63;
        sW2[n][k] = (_Float16)((n < GC) ? w2[n * HID + k] : 0.0f);
    }
    if (tid < 2 * GC) sGB[tid] = gb[b * (2 * GC) + tid];

#ifdef NCA_ASYNC_LDS
    async_wait0();
#endif
    __syncthreads();

    // ---- Phase B: depthwise 3x3 (cross-correlation), packed f16-pair stores ----
    {
        const int lx = tid & (TW - 1);
        const int ly = tid >> 5;
        #pragma unroll
        for (int op = 0; op < PCH / 2; ++op) {     // 18 channel pairs
            float acc0 = 0.0f, acc1 = 0.0f;
            const int oc0 = 2 * op, oc1 = 2 * op + 1;
            const int ic0 = oc0 / 3, ic1 = oc1 / 3;
            #pragma unroll
            for (int ky = 0; ky < 3; ++ky)
                #pragma unroll
                for (int kx = 0; kx < 3; ++kx) {
                    acc0 += sGrid[ic0][ly + ky][lx + kx] * sPw[oc0 * 9 + ky * 3 + kx];
                    acc1 += sGrid[ic1][ly + ky][lx + kx] * sPw[oc1 * 9 + ky * 3 + kx];
                }
            union { _Float16 f[2]; unsigned u; } pk;
            pk.f[0] = (_Float16)acc0;
            pk.f[1] = (_Float16)acc1;
            *(unsigned*)&sPerc[tid][2 * op] = pk.u;
        }
        #pragma unroll
        for (int k = PCH; k < KP; k += 2)
            *(unsigned*)&sPerc[tid][k] = 0u;
    }
    __syncthreads();

    const int wave = tid >> 5;
    const int lane = tid & 31;
    const int hf   = lane >> 4;      // lane half (0: lanes 0-15, 1: lanes 16-31)
    const int lr   = lane & 15;      // A-row (M) / B-col (N) within a fragment

    union FragU { v16h h; unsigned u[8]; };

    // ---- Phase C: GEMM1  hidden[256x64] = perc[256x64] x w1^T, +bias, ReLU ----
    {
        const int nT    = wave & 3;          // 16-wide out-channel tile, fixed per wave
        const int gBase = wave >> 2;         // pixel-group base
        const int nr    = nT * 16 + lr;
        FragU w0, w1f;
        #pragma unroll
        for (int p = 0; p < 8; ++p) {        // B layout: K = elem + 16*half
            const int kb = 2 * p + 16 * hf;
            w0.u[p]  = *(const unsigned*)&sW1[nr][kb];
            w1f.u[p] = *(const unsigned*)&sW1[nr][32 + kb];
        }
        const float bias = sB1[nr];
        #pragma unroll
        for (int i = 0; i < 8; ++i) {        // compile-time trip count: no exec masking
            const int g  = gBase + 2 * i;
            const int px = g * 16 + lr;
            FragU a0, a1;
            #pragma unroll
            for (int p = 0; p < 8; ++p) {    // A layout: K = elem + 16*half
                const int kb = 2 * p + 16 * hf;
                a0.u[p] = *(const unsigned*)&sPerc[px][kb];
                a1.u[p] = *(const unsigned*)&sPerc[px][32 + kb];
            }
            v8f c = {0.f, 0.f, 0.f, 0.f, 0.f, 0.f, 0.f, 0.f};
            c = __builtin_amdgcn_wmma_f32_16x16x32_f16(false, a0.h, false, w0.h,  (short)0, c, false, false);
            c = __builtin_amdgcn_wmma_f32_16x16x32_f16(false, a1.h, false, w1f.h, (short)0, c, false, false);
            #pragma unroll
            for (int r = 0; r < 8; ++r) {    // D layout: M = r + 8*half, N = lr
                const int m = r + 8 * hf;
                float hv = fmaxf(c[r] + bias, 0.0f);
                sHid[g * 16 + m][nT * 16 + lr] = (_Float16)hv;
            }
        }
    }
    __syncthreads();

    // ---- Phase D: GEMM2  upd[256x12] = hid[256x64] x w2^T ----
    {
        FragU w0, w1f;
        #pragma unroll
        for (int p = 0; p < 8; ++p) {
            const int kb = 2 * p + 16 * hf;
            w0.u[p]  = *(const unsigned*)&sW2[lr][kb];
            w1f.u[p] = *(const unsigned*)&sW2[lr][32 + kb];
        }
        #pragma unroll
        for (int i = 0; i < 2; ++i) {
            const int g  = wave + 8 * i;
            const int px = g * 16 + lr;
            FragU a0, a1;
            #pragma unroll
            for (int p = 0; p < 8; ++p) {
                const int kb = 2 * p + 16 * hf;
                a0.u[p] = *(const unsigned*)&sHid[px][kb];
                a1.u[p] = *(const unsigned*)&sHid[px][32 + kb];
            }
            v8f c = {0.f, 0.f, 0.f, 0.f, 0.f, 0.f, 0.f, 0.f};
            c = __builtin_amdgcn_wmma_f32_16x16x32_f16(false, a0.h, false, w0.h,  (short)0, c, false, false);
            c = __builtin_amdgcn_wmma_f32_16x16x32_f16(false, a1.h, false, w1f.h, (short)0, c, false, false);
            #pragma unroll
            for (int r = 0; r < 8; ++r) {
                const int m = r + 8 * hf;
                sUpd[g * 16 + m][lr] = c[r];
            }
        }
    }
    __syncthreads();

    // ---- Phase E: FiLM update + outputs (coalesced channel-major stores) ----
    {
        const int lx = tid & (TW - 1);
        const int ly = tid >> 5;
        const size_t pix = (size_t)(ty0 + ly) * WW + (tx0 + lx);
        float* gbo = gout + (size_t)b * GC * HH * WW;
        #pragma unroll
        for (int c = 0; c < GC; ++c) {
            const float oldv = sGrid[c][ly + 1][lx + 1];
            const float nv = oldv + sGB[c] * sUpd[tid][c] + sGB[GC + c];
            gbo[(size_t)c * HH * WW + pix] = nv;
            if (c == 0) {
                const float rgb = 1.0f / (1.0f + __expf(-nv));
                traj[(size_t)b * HH * WW + pix] = rgb;
                if (finalrgb) finalrgb[(size_t)b * HH * WW + pix] = rgb;
            }
        }
    }
}

// ---------------------------------------------------------------- launcher
extern "C" void kernel_launch(void* const* d_in, const int* in_sizes, int n_in,
                              void* d_out, int out_size, void* d_ws, size_t ws_size,
                              hipStream_t stream) {
    const int*   labels = (const int*)  d_in[0];
    const float* emb    = (const float*)d_in[1];
    const float* pw     = (const float*)d_in[2];
    const float* w1     = (const float*)d_in[3];
    const float* b1     = (const float*)d_in[4];
    const float* w2     = (const float*)d_in[5];
    const float* f1w    = (const float*)d_in[6];
    const float* f1b    = (const float*)d_in[7];
    const float* f2w    = (const float*)d_in[8];
    const float* f2b    = (const float*)d_in[9];
    // d_in[10] = num_steps (device scalar); fixed to 32 by out_size -> compile-time.

    float* out = (float*)d_out;

    const size_t GRID_ELEMS = (size_t)BATCH * GC * HH * WW;   // 12.58M floats
    float* bufA = (float*)d_ws;                                // step-even input
    float* bufB = bufA + GRID_ELEMS;                           // step-even output
    float* gbuf = bufB + GRID_ELEMS;                           // gamma/beta [B][24]

    nca_init_kernel<<<2048, 256, 0, stream>>>(bufA);
    nca_film_kernel<<<1, BATCH, 0, stream>>>(labels, emb, f1w, f1b, f2w, f2b, gbuf);

    dim3 grid(WW / TW, HH / TH, BATCH);                        // 4 x 16 x 64 blocks
    float* traj = out + (size_t)BATCH * HH * WW;               // traj after traj[-1] slot
    for (int s = 0; s < NSTEPS; ++s) {
        const float* gi = (s & 1) ? bufB : bufA;
        float*       go = (s & 1) ? bufA : bufB;
        float*       fr = (s == NSTEPS - 1) ? out : nullptr;
        nca_step_kernel<<<grid, 256, 0, stream>>>(
            gi, go, pw, w1, b1, w2, gbuf,
            traj + (size_t)s * BATCH * HH * WW, fr);
    }
}
